// MultiHeadAttention_46943992545302
// MI455X (gfx1250) — matrix operand order repaired and checked by reading, not compiled, not run on hardware
//
#include <hip/hip_runtime.h>
#include <hip/hip_bf16.h>

typedef __attribute__((ext_vector_type(16))) _Float16 v16h;
typedef __attribute__((ext_vector_type(8)))  _Float16 v8h;
typedef __attribute__((ext_vector_type(8)))  float    v8f;

#define DIMD   768
#define NHEAD  12
#define HD     64
#define BATCH  4
#define SEQ    2048
#define TOK    (BATCH * SEQ)     /* 8192 */
#define QKV_N  (3 * DIMD)        /* 2304 */

/* ---------- WMMA fragment loaders (wave32, gfx1250; A/B use contiguous k = 16h+s, not the ISA table order) ----------
 * A (16x32 f16): lane l holds row m=l&15; half i -> k = i + 16*(l>>4)
 *   => two contiguous 16B chunks at k0 + 16*(l>>4) and +8.
 * B (32x16 f16) given B^T row-major [Ncol][K]: lane l holds col n=l&15,
 *   16 contiguous K halves starting at k0 + 16*(l>>4)  => one 32B load.
 * C/D (16x16 f32): vgpr r, lane l -> element (m = 8*(l>>4)+r, n = l&15).
 */
static __device__ __forceinline__ v16h frag_a(const _Float16* base, int row0, int ldk,
                                              int k0, int lane) {
  int l = lane & 15, hi = lane >> 4;
  const _Float16* p = base + (size_t)(row0 + l) * ldk + k0 + hi * 16;
  v8h lo = *(const v8h*)p;
  v8h hh = *(const v8h*)(p + 8);
  v16h r;
#pragma unroll
  for (int i = 0; i < 8; ++i) { r[i] = lo[i]; r[i + 8] = hh[i]; }
  return r;
}

static __device__ __forceinline__ v16h frag_b(const _Float16* baseT, int col0, int ldk,
                                              int k0, int lane) {
  int l = lane & 15, hi = lane >> 4;
  return *(const v16h*)(baseT + (size_t)(col0 + l) * ldk + k0 + hi * 16);
}

static __device__ __forceinline__ v8f wmma_f16(v16h a, v16h b, v8f c) {
  return __builtin_amdgcn_wmma_f32_16x16x32_f16(false, a, false, b, (short)0, c,
                                                false, false);
}

/* ---------------- elementwise convert / transpose ---------------- */
__global__ void mha_cvt_f16(const float* __restrict__ a, _Float16* __restrict__ b, int n) {
  int i = blockIdx.x * blockDim.x + threadIdx.x;
  if (i < n) b[i] = (_Float16)a[i];
}

/* w: [K][Nc] fp32 -> wt: [Nc][K] f16 */
__global__ void mha_transpose_f16(const float* __restrict__ w, _Float16* __restrict__ wt,
                                  int K, int Nc) {
  int idx = blockIdx.x * blockDim.x + threadIdx.x;
  if (idx >= K * Nc) return;
  int n = idx / K, k = idx - n * K;
  wt[idx] = (_Float16)w[(size_t)k * Nc + n];
}

/* ------------- WMMA GEMM: C[M][Nc] = A[M][K](f16) * Bt[Nc][K](f16) + bias -------------
 * block = 256 threads = 8 waves laid out 4(M) x 2(N); wave tile 16x64; block tile 64x128.
 */
__global__ __launch_bounds__(256) void mha_wmma_gemm(const _Float16* __restrict__ A,
                                                     const _Float16* __restrict__ Bt,
                                                     const float* __restrict__ bias,
                                                     float* __restrict__ C,
                                                     int Nc, int K) {
  int wave = threadIdx.x >> 5, lane = threadIdx.x & 31;
  int l = lane & 15, hi = lane >> 4;
  int wm = wave & 3, wn = wave >> 2;
  int m0 = blockIdx.y * 64 + wm * 16;
  int n0 = blockIdx.x * 128 + wn * 64;

  v8f acc[4];
#pragma unroll
  for (int t = 0; t < 4; ++t) acc[t] = (v8f){};

  for (int k0 = 0; k0 < K; k0 += 32) {
    v16h af = frag_a(A, m0, K, k0, lane);
#pragma unroll
    for (int t = 0; t < 4; ++t) {
      v16h bf = frag_b(Bt, n0 + 16 * t, K, k0, lane);
      acc[t] = wmma_f16(af, bf, acc[t]);
    }
  }
#pragma unroll
  for (int t = 0; t < 4; ++t) {
    int n = n0 + 16 * t + l;
    float bv = bias[n];
#pragma unroll
    for (int r = 0; r < 8; ++r) {
      int m = m0 + 8 * hi + r;
      C[(size_t)m * Nc + n] = acc[t][r] + bv;
    }
  }
}

/* ------------- QKV epilogue: RMSNorm(q,k) + RoPE, split heads -------------
 * 1 wave per (b,h,n) row; lane j handles dims j and j+32.
 * qh/kh: [B][H][N][64] f16 ; vt: [B][H][64][N] f16 (transposed for PV WMMA B operand)
 */
__global__ __launch_bounds__(256) void mha_qkv_post(const float* __restrict__ qkv,
                                                    const float* __restrict__ qw,
                                                    const float* __restrict__ kw,
                                                    _Float16* __restrict__ qh,
                                                    _Float16* __restrict__ kh,
                                                    _Float16* __restrict__ vt) {
  int wave = threadIdx.x >> 5, j = threadIdx.x & 31;
  int row = blockIdx.x * 8 + wave;            /* over B*H*N */
  int b = row / (NHEAD * SEQ);
  int rem = row - b * (NHEAD * SEQ);
  int h = rem / SEQ;
  int n = rem - h * SEQ;

  size_t token = (size_t)b * SEQ + n;
  const float* qp = qkv + token * QKV_N + h * HD;
  const float* kp = qp + DIMD;
  const float* vp = qp + 2 * DIMD;

  float q0 = qp[j], q1 = qp[j + 32];
  float k0 = kp[j], k1 = kp[j + 32];

  float sq = q0 * q0 + q1 * q1;
  float sk = k0 * k0 + k1 * k1;
#pragma unroll
  for (int off = 1; off < 32; off <<= 1) {
    sq += __shfl_xor(sq, off, 32);
    sk += __shfl_xor(sk, off, 32);
  }
  float rq = rsqrtf(sq * (1.0f / 64.0f) + 1e-6f);
  float rk = rsqrtf(sk * (1.0f / 64.0f) + 1e-6f);
  q0 *= rq * qw[j];  q1 *= rq * qw[j + 32];
  k0 *= rk * kw[j];  k1 *= rk * kw[j + 32];

  /* RoPE: inv_freq[j] = theta^(-j/32); out = t*cos + rotate_half(t)*sin */
  float invf = __powf(10000.0f, -(float)j * (1.0f / 32.0f));
  float ang = (float)n * invf;
  float c = cosf(ang), s = sinf(ang);
  float qo0 = q0 * c - q1 * s, qo1 = q1 * c + q0 * s;
  float ko0 = k0 * c - k1 * s, ko1 = k1 * c + k0 * s;

  size_t bh = (size_t)b * NHEAD + h;
  size_t rbase = (bh * SEQ + n) * HD;
  qh[rbase + j]      = (_Float16)qo0;
  qh[rbase + j + 32] = (_Float16)qo1;
  kh[rbase + j]      = (_Float16)ko0;
  kh[rbase + j + 32] = (_Float16)ko1;

  size_t vbase = bh * (size_t)HD * SEQ;
  vt[vbase + (size_t)j * SEQ + n]        = (_Float16)vp[j];
  vt[vbase + (size_t)(j + 32) * SEQ + n] = (_Float16)vp[j + 32];
}

/* ------------- Flash attention: 1 wave per 16 query rows, 32 keys/step -------------
 * ah: [TOK][DIMD] f16, col offset h*64.
 */
__global__ __launch_bounds__(256) void mha_flash(const _Float16* __restrict__ qh,
                                                 const _Float16* __restrict__ kh,
                                                 const _Float16* __restrict__ vt,
                                                 _Float16* __restrict__ ah) {
  __shared__ _Float16 plds[8 * 16 * 32];
  int wave = threadIdx.x >> 5, lane = threadIdx.x & 31;
  int l = lane & 15, hi = lane >> 4;

  const int qchunks = SEQ / 128;
  int bh = blockIdx.x / qchunks;
  int q0 = (blockIdx.x - bh * qchunks) * 128 + wave * 16;
  int b = bh / NHEAD, h = bh - b * NHEAD;

  const _Float16* Q = qh + (size_t)bh * SEQ * HD;
  const _Float16* K = kh + (size_t)bh * SEQ * HD;
  const _Float16* V = vt + (size_t)bh * HD * SEQ;
  _Float16* pbuf = plds + wave * (16 * 32);

  v16h qf0 = frag_a(Q, q0, HD, 0, lane);
  v16h qf1 = frag_a(Q, q0, HD, 32, lane);

  float mrow[8], lrow[8];
  v8f o0 = (v8f){}, o1 = (v8f){}, o2 = (v8f){}, o3 = (v8f){};
#pragma unroll
  for (int r = 0; r < 8; ++r) { mrow[r] = -1e30f; lrow[r] = 0.0f; }
  const float scale = 0.125f; /* 1/sqrt(64) */

  for (int kb = 0; kb < SEQ; kb += 32) {
    /* S = Q K^T : key tiles [kb..kb+15] and [kb+16..kb+31], K-dim = 64 */
    v8f s0 = (v8f){}, s1 = (v8f){};
#pragma unroll
    for (int c = 0; c < 2; ++c) {
      v16h kf0 = frag_b(K, kb, HD, c * 32, lane);
      v16h kf1 = frag_b(K, kb + 16, HD, c * 32, lane);
      v16h qa = (c == 0) ? qf0 : qf1;
      s0 = wmma_f16(qa, kf0, s0);
      s1 = wmma_f16(qa, kf1, s1);
    }
    /* online softmax over the 32 new key columns */
#pragma unroll
    for (int r = 0; r < 8; ++r) {
      float a = s0[r] * scale, bb = s1[r] * scale;
      float mx = fmaxf(a, bb);
#pragma unroll
      for (int off = 1; off < 16; off <<= 1) mx = fmaxf(mx, __shfl_xor(mx, off, 32));
      float mnew = fmaxf(mrow[r], mx);
      float corr = __expf(mrow[r] - mnew);
      float p0 = __expf(a - mnew), p1 = __expf(bb - mnew);
      float rs = p0 + p1;
#pragma unroll
      for (int off = 1; off < 16; off <<= 1) rs += __shfl_xor(rs, off, 32);
      lrow[r] = lrow[r] * corr + rs;
      mrow[r] = mnew;
      o0[r] *= corr; o1[r] *= corr; o2[r] *= corr; o3[r] *= corr;
      int prow = 8 * hi + r;
      pbuf[prow * 32 + l]      = (_Float16)p0;
      pbuf[prow * 32 + 16 + l] = (_Float16)p1;
    }
    /* re-shape P (C layout) -> A fragment via per-wave LDS */
    v16h pf = frag_a(pbuf, 0, 32, 0, lane);
    /* O += P * V : B^T = vt rows are output dims d, contiguous over keys */
    v16h vf0 = frag_b(V, 0,  SEQ, kb, lane);
    v16h vf1 = frag_b(V, 16, SEQ, kb, lane);
    v16h vf2 = frag_b(V, 32, SEQ, kb, lane);
    v16h vf3 = frag_b(V, 48, SEQ, kb, lane);
    o0 = wmma_f16(pf, vf0, o0);
    o1 = wmma_f16(pf, vf1, o1);
    o2 = wmma_f16(pf, vf2, o2);
    o3 = wmma_f16(pf, vf3, o3);
  }

#pragma unroll
  for (int r = 0; r < 8; ++r) {
    float inv = 1.0f / lrow[r];
    int token = b * SEQ + q0 + 8 * hi + r;
    size_t rb = (size_t)token * DIMD + h * HD;
    ah[rb + l]      = (_Float16)(o0[r] * inv);
    ah[rb + 16 + l] = (_Float16)(o1[r] * inv);
    ah[rb + 32 + l] = (_Float16)(o2[r] * inv);
    ah[rb + 48 + l] = (_Float16)(o3[r] * inv);
  }
}

/* ---------------- launch ---------------- */
extern "C" void kernel_launch(void* const* d_in, const int* in_sizes, int n_in,
                              void* d_out, int out_size, void* d_ws, size_t ws_size,
                              hipStream_t stream) {
  const float* x        = (const float*)d_in[0];
  const float* qkv_w    = (const float*)d_in[1];
  const float* qkv_b    = (const float*)d_in[2];
  const float* proj_w   = (const float*)d_in[3];
  const float* proj_b   = (const float*)d_in[4];
  const float* q_norm_w = (const float*)d_in[5];
  const float* k_norm_w = (const float*)d_in[6];
  float* out = (float*)d_out;

  char* ws = (char*)d_ws;
  _Float16* xh     = (_Float16*)(ws + 0);          /* 8192*768   f16 = 12,582,912 */
  _Float16* wqkvT  = (_Float16*)(ws + 12582912);   /* 2304*768   f16 =  3,538,944 */
  _Float16* wprojT = (_Float16*)(ws + 16121856);   /* 768*768    f16 =  1,179,648 */
  float*    qkv32  = (float*)   (ws + 17301504);   /* 8192*2304  f32 = 75,497,472 */
  _Float16* qh     = (_Float16*)(ws + 92798976);   /* 48*2048*64 f16 = 12,582,912 */
  _Float16* kh     = (_Float16*)(ws + 105381888);
  _Float16* vt     = (_Float16*)(ws + 117964800);
  _Float16* ah     = (_Float16*)(ws + 130547712);  /* end: 143,130,624 bytes */

  mha_cvt_f16<<<(TOK * DIMD) / 256, 256, 0, stream>>>(x, xh, TOK * DIMD);
  mha_transpose_f16<<<(DIMD * QKV_N + 255) / 256, 256, 0, stream>>>(qkv_w, wqkvT, DIMD, QKV_N);
  mha_transpose_f16<<<(DIMD * DIMD + 255) / 256, 256, 0, stream>>>(proj_w, wprojT, DIMD, DIMD);

  mha_wmma_gemm<<<dim3(QKV_N / 128, TOK / 64), 256, 0, stream>>>(xh, wqkvT, qkv_b, qkv32,
                                                                 QKV_N, DIMD);
  mha_qkv_post<<<(BATCH * NHEAD * SEQ) / 8, 256, 0, stream>>>(qkv32, q_norm_w, k_norm_w,
                                                              qh, kh, vt);
  mha_flash<<<BATCH * NHEAD * (SEQ / 128), 256, 0, stream>>>(qh, kh, vt, ah);
  mha_wmma_gemm<<<dim3(DIMD / 128, TOK / 64), 256, 0, stream>>>(ah, wprojT, proj_b, out,
                                                                DIMD, DIMD);
}
